// MoE_25409026523791
// MI455X (gfx1250) — matrix operand order repaired and checked by reading, not compiled, not run on hardware
//
#include <hip/hip_runtime.h>
#include <hip/hip_bf16.h>

// ---------------------------------------------------------------------------
//   x: (4,2048,1024) f32 -> T = 8192 tokens, D = 1024
//   out[t] = (silu(x[t] @ up^T) @ down^T) * top2scale(t)
// Pipeline:
//   k0: f32 -> bf16 pre-convert (x, up_w, down_w)        [memory bound]
//   k1: gate scales (one wave32/token)
//   k2: H = silu(x_bf @ up_bf^T)  bf16   (TDM-staged WMMA GEMM)
//   k3: out = (H @ down_bf^T) * scale    (TDM-staged WMMA GEMM)
// ---------------------------------------------------------------------------

#define D_DIM   1024
#define ED_DIM  4096
#define NE_DIM  8

typedef __attribute__((ext_vector_type(16))) __bf16    v16bf;
typedef __attribute__((ext_vector_type(8)))  float     v8f;
typedef __attribute__((ext_vector_type(4)))  unsigned  v4u;
typedef __attribute__((ext_vector_type(8)))  int       v8i;
typedef __attribute__((ext_vector_type(4)))  int       v4i;

union BFrag {
    v16bf    v;
    unsigned u[8];
};

__device__ __forceinline__ unsigned f2bf(float f) {
    unsigned x = __float_as_uint(f);
    unsigned r = x + 0x7FFFu + ((x >> 16) & 1u);   // RNE bf16
    return r >> 16;
}
__device__ __forceinline__ unsigned pack2bf(float lo, float hi) {
    return f2bf(lo) | (f2bf(hi) << 16);
}

// ---------------------------------------------------------------------------
// Kernel 0: f32 -> bf16 conversion
// ---------------------------------------------------------------------------
__global__ __launch_bounds__(256) void cvt_f32_to_bf16_kernel(
    const float4* __restrict__ in, uint2* __restrict__ out, int n4)
{
    int i = blockIdx.x * blockDim.x + threadIdx.x;
    if (i >= n4) return;
    float4 v = in[i];
    uint2 p;
    p.x = pack2bf(v.x, v.y);
    p.y = pack2bf(v.z, v.w);
    out[i] = p;
}

// ---------------------------------------------------------------------------
// Kernel 1: gating. One wave32 per token -> scale[t] = s2/(s2+1e-9)
// ---------------------------------------------------------------------------
__global__ __launch_bounds__(256) void moe_gate_kernel(
    const float* __restrict__ x, const float* __restrict__ gate_w,
    float* __restrict__ scale, int T)
{
    int gtid = blockIdx.x * blockDim.x + threadIdx.x;
    int wave = gtid >> 5;
    int lane = threadIdx.x & 31;
    if (wave >= T) return;

    const float* xr = x + (size_t)wave * D_DIM;
    float acc[NE_DIM];
#pragma unroll
    for (int e = 0; e < NE_DIM; ++e) acc[e] = 0.0f;

    for (int i = lane; i < D_DIM; i += 32) {
        float xv = xr[i];
#pragma unroll
        for (int e = 0; e < NE_DIM; ++e)
            acc[e] += xv * gate_w[e * D_DIM + i];
    }
#pragma unroll
    for (int off = 16; off > 0; off >>= 1)
#pragma unroll
        for (int e = 0; e < NE_DIM; ++e)
            acc[e] += __shfl_xor(acc[e], off, 32);

    if (lane == 0) {
        float mx = acc[0];
#pragma unroll
        for (int e = 1; e < NE_DIM; ++e) mx = fmaxf(mx, acc[e]);
        float sum = 0.0f;
#pragma unroll
        for (int e = 0; e < NE_DIM; ++e) sum += __expf(acc[e] - mx);
        float a = -3.4e38f, b = -3.4e38f;
#pragma unroll
        for (int e = 0; e < NE_DIM; ++e) {
            float v = acc[e];
            if (v > a)      { b = a; a = v; }
            else if (v > b) { b = v; }
        }
        float s2 = (__expf(a - mx) + __expf(b - mx)) / sum;
        scale[wave] = s2 / (s2 + 1e-9f);
    }
}

// ---------------------------------------------------------------------------
// TDM tile load: one 2D descriptor moves a [128 rows x 64 cols] bf16 tile
// (row stride K elements) from global into LDS, inserting a 1-dword pad after
// every 128-byte row => LDS row stride 132B = 33 banks (conflict-free).
//   D# group0: count=1, lds_addr, 57b global byte addr, type=2
//   D# group1: data_size=2B, pad_enable, pad_interval=32dw, pad_amount=1dw,
//              tensor_dim0=K, tensor_dim1=rows, tile=64x128, dim0_stride=K
// ---------------------------------------------------------------------------
__device__ __forceinline__ void tdm_load_tile(
    const unsigned short* gtile,  // element pointer to tile start
    unsigned lds_off,             // LDS byte offset of tile dest
    int rows, int K)
{
    unsigned long long ga = (unsigned long long)(uintptr_t)gtile;

    v4u g0;
    g0[0] = 1u;                                        // count=1, user mode
    g0[1] = lds_off;                                   // lds_addr
    g0[2] = (unsigned)(ga & 0xFFFFFFFFu);              // global_addr[31:0]
    g0[3] = (unsigned)((ga >> 32) & 0x01FFFFFFu)       // global_addr[56:32]
          | (2u << 30);                                // type=2 (image)

    v8i g1;
    g1[0] = (int)((1u << 16)                           // data_size = 2 bytes
          |       (1u << 20)                           // pad_enable
          |       (4u << 22));                         // pad_interval: 32 dwords
                                                       // pad_amount=0 -> 1 dword
    g1[1] = (int)(((unsigned)K & 0xFFFFu) << 16);      // tensor_dim0 lo16
    g1[2] = (int)((((unsigned)K) >> 16)                // tensor_dim0 hi16
          | (((unsigned)rows & 0xFFFFu) << 16));       // tensor_dim1 lo16
    g1[3] = (int)((((unsigned)rows) >> 16)             // tensor_dim1 hi16
          | (64u << 16));                              // tile_dim0 = 64
    g1[4] = 128;                                       // tile_dim1=128, tile_dim2=0
    g1[5] = K;                                         // tensor_dim0_stride lo32
    g1[6] = 0;
    g1[7] = 0;

    v4i z4 = {0, 0, 0, 0};
    v8i z8 = {0, 0, 0, 0, 0, 0, 0, 0};
    // amdgpu-toolchain (clang-23) 6-arg form:
    //   (uint32x4 g0, int32x8 g1, int32x4, int32x4, int32x8, i32 cpol)
    __builtin_amdgcn_tensor_load_to_lds(g0, g1, z4, z4, z8, 0);
}

// ---------------------------------------------------------------------------
// TDM-staged, double-buffered bf16 WMMA GEMM:  C(MxN) = A(MxK) * W(NxK)^T
//   Block tile 128x128, 8 wave32, wave subtile 32x64 (2x4 of 16x16x32 WMMA)
//   K chunks of 64; wave0 DMAs chunk kc+1 while all waves compute chunk kc.
// ---------------------------------------------------------------------------
#define LDA        66                       // shorts per LDS row (132 B)
#define TILE_BYTES (128 * 132)              // 16896
#define OFF_A0     0
#define OFF_A1     (TILE_BYTES)
#define OFF_B0     (2 * TILE_BYTES)
#define OFF_B1     (3 * TILE_BYTES)
#define SMEM_BYTES (4 * TILE_BYTES)         // 67584

template <bool SILU_BF16_OUT>
__global__ __launch_bounds__(256) void ffn_gemm_kernel(
    const unsigned short* __restrict__ A,    // [M][K] bf16
    const unsigned short* __restrict__ W,    // [N][K] bf16
    void*  __restrict__ Out,                 // bf16 H (silu) or f32 (scaled)
    const float* __restrict__ scale,         // [M] row scale (f32 path)
    int M, int N, int K)
{
    extern __shared__ char smem[];

    const int tid  = threadIdx.x;
    const int bm   = blockIdx.y;
    const int bn   = blockIdx.x;
    const int wave = tid >> 5;
    const int lane = tid & 31;
    const int wm   = wave & 3;    // 4 waves along M (32 rows each)
    const int wn   = wave >> 2;   // 2 waves along N (64 cols each)
    const int half = lane >> 4;
    const int r    = lane & 15;

    const unsigned short* Abase = A + (size_t)(bm * 128) * K;
    const unsigned short* Wbase = W + (size_t)(bn * 128) * K;

    v8f acc[2][4];
#pragma unroll
    for (int mt = 0; mt < 2; ++mt)
#pragma unroll
        for (int nt = 0; nt < 4; ++nt) {
            v8f z = {};
            acc[mt][nt] = z;
        }

    const int nChunks = K >> 6;

    // ---- prologue: DMA chunk 0 into buffer 0 ----
    if (wave == 0) {
        tdm_load_tile(Abase, OFF_A0, M, K);
        tdm_load_tile(Wbase, OFF_B0, N, K);
    }

    int p = 0;
    for (int kc = 0; kc < nChunks; ++kc) {
        // ---- wave0: DMA next chunk into the other buffer, then wait until
        //      the current chunk's two TDM ops have landed (in-order cnt) ----
        if (wave == 0) {
            if (kc + 1 < nChunks) {
                int kb2 = (kc + 1) << 6;
                tdm_load_tile(Abase + kb2, p ? OFF_A0 : OFF_A1, M, K);
                tdm_load_tile(Wbase + kb2, p ? OFF_B0 : OFF_B1, N, K);
                __builtin_amdgcn_s_wait_tensorcnt(2);
            } else {
                __builtin_amdgcn_s_wait_tensorcnt(0);
            }
        }
        __syncthreads();   // current chunk visible to all waves

        const unsigned short* lsA = (const unsigned short*)(smem + (p ? OFF_A1 : OFF_A0));
        const unsigned short* lsB = (const unsigned short*)(smem + (p ? OFF_B1 : OFF_B0));

#pragma unroll
        for (int kk = 0; kk < 64; kk += 32) {
            BFrag afr[2], bfr[4];
            // A 16x32 bf16 fragment (ISA layout):
            //   lanes 0-15: row=r, K {0..7,16..23}; lanes 16-31: K {8..15,24..31}
#pragma unroll
            for (int mt = 0; mt < 2; ++mt) {
                int row = wm * 32 + mt * 16 + r;
                const unsigned short* base = &lsA[row * LDA + kk];
#pragma unroll
                for (int j = 0; j < 8; ++j) {
                    int ko = ((j & 4) << 2) + half * 8 + ((j & 3) << 1);
                    afr[mt].u[j] = *(const unsigned*)(base + ko);
                }
            }
            // B 32x16 bf16 fragment: lanes 0-15 col=r K {0..7,16..23}; 16-31 K {8..15,24..31}
#pragma unroll
            for (int nt = 0; nt < 4; ++nt) {
                int col = wn * 64 + nt * 16 + r;
                const unsigned short* base = &lsB[col * LDA + kk + half * 8];
#pragma unroll
                for (int j = 0; j < 8; ++j)
                    bfr[nt].u[j] = *(const unsigned*)(base + ((j & 4) << 2) + ((j & 3) << 1));
            }
#pragma unroll
            for (int mt = 0; mt < 2; ++mt)
#pragma unroll
                for (int nt = 0; nt < 4; ++nt)
                    acc[mt][nt] = __builtin_amdgcn_wmma_f32_16x16x32_bf16(
                        false, afr[mt].v, false, bfr[nt].v,
                        (short)0, acc[mt][nt], false, false);
        }
        __syncthreads();   // all waves done with buffer p; safe to overwrite
        p ^= 1;
    }

    // ---- epilogue: C/D layout VGPR j -> row half*8+j, lane r -> col ----
#pragma unroll
    for (int mt = 0; mt < 2; ++mt) {
#pragma unroll
        for (int nt = 0; nt < 4; ++nt) {
            int col = bn * 128 + wn * 64 + nt * 16 + r;
#pragma unroll
            for (int j = 0; j < 8; ++j) {
                int row = bm * 128 + wm * 32 + mt * 16 + half * 8 + j;
                float v = acc[mt][nt][j];
                if (SILU_BF16_OUT) {
                    float s = v / (1.0f + __expf(-v));
                    ((unsigned short*)Out)[(size_t)row * N + col] =
                        (unsigned short)f2bf(s);
                } else {
                    ((float*)Out)[(size_t)row * N + col] = v * scale[row];
                }
            }
        }
    }
}

// ---------------------------------------------------------------------------
// Launcher
// ---------------------------------------------------------------------------
extern "C" void kernel_launch(void* const* d_in, const int* in_sizes, int n_in,
                              void* d_out, int out_size, void* d_ws, size_t ws_size,
                              hipStream_t stream)
{
    const float* x      = (const float*)d_in[0];  // [T][1024]
    const float* gate_w = (const float*)d_in[1];  // [8][1024]
    const float* up_w   = (const float*)d_in[2];  // [4096][1024]
    const float* down_w = (const float*)d_in[3];  // [1024][4096]
    float*       out    = (float*)d_out;          // [T][1024]

    const int T = in_sizes[0] / D_DIM;            // 8192

    // workspace layout (bytes):
    //   [0, 32K)        scale f32 [T]
    //   [32K, +16M)     x_bf16   [T][1024]
    //   next 8M         up_bf16  [4096][1024]
    //   next 8M         down_bf16[1024][4096]
    //   next 64M        H bf16   [T][4096]
    char* ws = (char*)d_ws;
    float*          ws_scale = (float*)ws;
    unsigned short* ws_xbf   = (unsigned short*)(ws + 32768);
    unsigned short* ws_upbf  = ws_xbf   + (size_t)T * D_DIM;
    unsigned short* ws_dnbf  = ws_upbf  + (size_t)ED_DIM * D_DIM;
    unsigned short* ws_H     = ws_dnbf  + (size_t)D_DIM * ED_DIM;

    // 0) pre-convert inputs/weights to bf16
    {
        int n4x = (T * D_DIM) / 4;
        cvt_f32_to_bf16_kernel<<<(n4x + 255) / 256, 256, 0, stream>>>(
            (const float4*)x, (uint2*)ws_xbf, n4x);
        int n4w = (ED_DIM * D_DIM) / 4;
        cvt_f32_to_bf16_kernel<<<(n4w + 255) / 256, 256, 0, stream>>>(
            (const float4*)up_w, (uint2*)ws_upbf, n4w);
        cvt_f32_to_bf16_kernel<<<(n4w + 255) / 256, 256, 0, stream>>>(
            (const float4*)down_w, (uint2*)ws_dnbf, n4w);
    }

    // 1) gating scales
    {
        int blocks = (T + 7) / 8;   // 8 wave32 per 256-thread block
        moe_gate_kernel<<<blocks, 256, 0, stream>>>(x, gate_w, ws_scale, T);
    }

    // 2) H = silu(x @ up^T)   (M=T, N=4096, K=1024)
    {
        dim3 grid(ED_DIM / 128, T / 128);
        ffn_gemm_kernel<true><<<grid, 256, SMEM_BYTES, stream>>>(
            ws_xbf, ws_upbf, (void*)ws_H, nullptr, T, ED_DIM, D_DIM);
    }

    // 3) out = (H @ down^T) * scale[row]   (M=T, N=1024, K=4096)
    {
        dim3 grid(D_DIM / 128, T / 128);
        ffn_gemm_kernel<false><<<grid, 256, SMEM_BYTES, stream>>>(
            ws_H, ws_dnbf, (void*)out, ws_scale, T, D_DIM, ED_DIM);
    }
}
